// Decoder_34351148434122
// MI455X (gfx1250) — matrix operand order repaired and checked by reading, not compiled, not run on hardware
//
#include <hip/hip_runtime.h>

// ---------------------------------------------------------------- types
typedef __attribute__((ext_vector_type(16))) __bf16 bf16x16;
typedef __attribute__((ext_vector_type(8)))  __bf16 bf16x8;
typedef __attribute__((ext_vector_type(8)))  float  floatx8;

union Frag16 { bf16x16 v; bf16x8 h[2]; };

static __device__ __forceinline__ __bf16 f2bf(float f) {
  union { float f; unsigned u; } in; in.f = f;
  unsigned u = in.u;
  u += 0x7fffu + ((u >> 16) & 1u);                 // round-to-nearest-even
  union { unsigned short s; __bf16 b; } out; out.s = (unsigned short)(u >> 16);
  return out.b;
}
static __device__ __forceinline__ float bf2f(__bf16 b) {
  union { __bf16 b; unsigned short s; } in; in.b = b;
  union { unsigned u; float f; } out; out.u = (unsigned)in.s << 16;
  return out.f;
}

// CDNA5 async global->LDS DMA (ASYNCcnt-tracked), per CDNA5 ISA.
// vdst = 32-bit LDS byte address, vaddr = 64-bit global address, 16B payload.
static __device__ __forceinline__ void async_copy_b128(void* lds, const void* g) {
  unsigned l = (unsigned)(size_t)lds;   // generic shared ptr: low 32 bits = LDS offset
  asm volatile("global_load_async_to_lds_b128 %0, %1, off" :: "v"(l), "v"(g) : "memory");
}
static __device__ __forceinline__ void wait_async0() {
  asm volatile("s_wait_asynccnt 0x0" ::: "memory");
}

// problem constants
#define CB   2
#define CS   2048
#define CHID 2048
#define CNH  16
#define CNKV 4
#define CHD  128
#define CFF  8192
#define CTOK (CB*CS)     // 4096

// ------------------------------------------------- weight convert + transpose
// in: K x N fp32 (row stride ldin, column offset col0) -> out: N x K bf16
__global__ __launch_bounds__(256)
void convert_transpose(const float* __restrict__ in, __bf16* __restrict__ out,
                       int K, int N, int ldin, int col0) {
  __shared__ float tile[32][33];
  const int bk = blockIdx.x * 32;
  const int bn = blockIdx.y * 32;
  const int tx = threadIdx.x & 31, ty = threadIdx.x >> 5;   // ty: 0..7
#pragma unroll
  for (int i = 0; i < 32; i += 8)
    tile[ty + i][tx] = in[(size_t)(bk + ty + i) * ldin + col0 + bn + tx];
  __syncthreads();
#pragma unroll
  for (int i = 0; i < 32; i += 8)
    out[(size_t)(bn + ty + i) * K + bk + tx] = f2bf(tile[tx][ty + i]);
}

// ---------------------------------------------------------------- RMSNorm -> bf16
__global__ __launch_bounds__(256)
void rmsnorm_kernel(const float* __restrict__ x, const float* __restrict__ w,
                    __bf16* __restrict__ out) {
  __shared__ float red[256];
  const int row = blockIdx.x;
  const float* xr = x + (size_t)row * CHID;
  float s = 0.f;
  for (int i = threadIdx.x; i < CHID; i += 256) { float v = xr[i]; s += v * v; }
  red[threadIdx.x] = s; __syncthreads();
  for (int st = 128; st > 0; st >>= 1) {
    if (threadIdx.x < st) red[threadIdx.x] += red[threadIdx.x + st];
    __syncthreads();
  }
  const float inv = rsqrtf(red[0] * (1.0f / CHID) + 1e-6f);
  for (int i = threadIdx.x; i < CHID; i += 256)
    out[(size_t)row * CHID + i] = f2bf(xr[i] * inv * w[i]);
}

// ------------------------------------------------- WMMA GEMM: C = A(MxK) * Bt^T
// A: MxK bf16 row-major. Bt: NxK bf16 row-major (pre-transposed weights).
// Block tile 128x128, 8 waves each 32x64. Double-buffered async LDS staging.
// EPI: 0 = f32 store, 1 = f32 + f32 aux (residual), 2 = bf16 silu(acc)*bf16 aux,
//      3 = bf16 store
#define SA_S 48

template<int EPI>
__global__ __launch_bounds__(256)
void gemm_wmma(const __bf16* __restrict__ A, const __bf16* __restrict__ Bt,
               void* __restrict__ Cout, const void* __restrict__ auxp,
               int M, int N, int K, int lda, int ldc) {
  __shared__ __bf16 sA[2][128 * SA_S];
  __shared__ __bf16 sB[2][128 * SA_S];
  const int tid  = threadIdx.x;
  const int wave = tid >> 5;
  const int lane = tid & 31;
  const int half = lane >> 4;
  const int l16  = lane & 15;
  const int bm0 = blockIdx.y * 128;
  const int bn0 = blockIdx.x * 128;
  const int wm0 = (wave >> 1) * 32;
  const int wn0 = (wave & 1) * 64;

  floatx8 acc[2][4];
  const floatx8 zf = {0.f,0.f,0.f,0.f,0.f,0.f,0.f,0.f};
#pragma unroll
  for (int i = 0; i < 2; i++)
#pragma unroll
    for (int j = 0; j < 4; j++) acc[i][j] = zf;

  const int kbaseA = half * 8;
  const int kbaseB = half * 8;

  auto stage = [&](int buf, int k0) {
#pragma unroll
    for (int t = 0; t < 2; t++) {
      const int c = tid + t * 256;          // 512 16B chunks per operand
      const int r = c >> 2, ck = c & 3;     // r: tile row, ck: 16B chunk in row
      async_copy_b128(&sA[buf][r * SA_S + ck * 8],
                      A + (size_t)(bm0 + r) * lda + k0 + ck * 8);
      async_copy_b128(&sB[buf][r * SA_S + ck * 8],
                      Bt + (size_t)(bn0 + r) * K + k0 + ck * 8);
    }
  };

  const int KT = K >> 5;
  stage(0, 0);
  wait_async0();
  __syncthreads();

  for (int kb = 0; kb < KT; kb++) {
    const int cur = kb & 1;
    if (kb + 1 < KT) stage(cur ^ 1, (kb + 1) * 32);   // overlap DMA with WMMA

    Frag16 af[2];
#pragma unroll
    for (int i = 0; i < 2; i++) {
      const __bf16* p = &sA[cur][(wm0 + i * 16 + l16) * SA_S + kbaseA];
      af[i].h[0] = *(const bf16x8*)p;
      af[i].h[1] = *(const bf16x8*)(p + 16);
    }
    Frag16 bfm[4];
#pragma unroll
    for (int j = 0; j < 4; j++) {
      const __bf16* p = &sB[cur][(wn0 + j * 16 + l16) * SA_S + kbaseB];
      bfm[j].h[0] = *(const bf16x8*)p;
      bfm[j].h[1] = *(const bf16x8*)(p + 16);
    }
#pragma unroll
    for (int i = 0; i < 2; i++)
#pragma unroll
      for (int j = 0; j < 4; j++)
        acc[i][j] = __builtin_amdgcn_wmma_f32_16x16x32_bf16(
            false, af[i].v, false, bfm[j].v, (short)0, acc[i][j], false, false);

    wait_async0();
    __syncthreads();
  }

#pragma unroll
  for (int i = 0; i < 2; i++)
#pragma unroll
    for (int j = 0; j < 4; j++) {
      const int col = bn0 + wn0 + j * 16 + l16;
#pragma unroll
      for (int r = 0; r < 8; r++) {
        const int row = bm0 + wm0 + i * 16 + r + half * 8;
        const size_t idx = (size_t)row * ldc + col;
        const float v = acc[i][j][r];
        if (EPI == 0) {
          ((float*)Cout)[idx] = v;
        } else if (EPI == 1) {
          ((float*)Cout)[idx] = v + ((const float*)auxp)[idx];
        } else if (EPI == 2) {
          const float g = v / (1.f + __expf(-v));
          ((__bf16*)Cout)[idx] = f2bf(g * bf2f(((const __bf16*)auxp)[idx]));
        } else {
          ((__bf16*)Cout)[idx] = f2bf(v);
        }
      }
    }
}

// ---------------------------------------------------- rope on v + bf16 pack
// kproj/vproj: (B,S,NKV,HD) fp32 -> kbf/vbf: (B,NKV,S,HD) bf16,
// vbfT: (B,NKV,HD,S) bf16 (pre-transposed for async V^T staging in flash).
__global__ __launch_bounds__(256)
void rope_pack(const float* __restrict__ kproj, const float* __restrict__ vproj,
               const float* __restrict__ cosb, const float* __restrict__ sinb,
               __bf16* __restrict__ kbf, __bf16* __restrict__ vbf,
               __bf16* __restrict__ vbfT) {
  const int idx = blockIdx.x * 256 + threadIdx.x;   // B*S*NKV*HD = 2097152
  const int d = idx & 127;
  const int g = (idx >> 7) & 3;
  const int s = (idx >> 9) & 2047;
  const int b = idx >> 20;
  const size_t in  = idx;
  const size_t out = ((((size_t)b * CNKV + g) * CS + s) * CHD + d);
  kbf[out] = f2bf(kproj[in]);
  float v = vproj[in];
  if (d < 64) {
    const size_t ci = ((size_t)b * CS + s) * 64 + d;
    const float c = cosb[ci], sn = sinb[ci];
    const float other = (d < 32) ? -vproj[in + 32] : vproj[in - 32];
    v = v * c + other * sn;
  }
  const __bf16 vb = f2bf(v);
  vbf[out] = vb;
  vbfT[((((size_t)b * CNKV + g) * CHD + d) * CS + s)] = vb;
}

// ---------------------------------------------------- flash attention (per kv head)
// Q = k[b,g], Keys = Values = v[b,g] (reference einsum). Causal, scale = sqrt(HD).
// One wave per 16 query rows, 4 waves/block share async-staged 32-key tiles.
__global__ __launch_bounds__(128)
void flash_attn(const __bf16* __restrict__ kbf, const __bf16* __restrict__ vbf,
                const __bf16* __restrict__ vbfT, float* __restrict__ attn_out) {
  __shared__ __bf16 sK[32 * 136];    // [key][d]
  __shared__ __bf16 sVT[128 * 48];   // [d][key]
  __shared__ __bf16 sP[4 * 16 * 48]; // per-wave P tile [row][key]
  const int tid  = threadIdx.x;
  const int wave = tid >> 5;
  const int lane = tid & 31;
  const int half = lane >> 4;
  const int l16  = lane & 15;
  const int bg  = blockIdx.x;              // b*NKV+g
  const int rb0 = blockIdx.y * 64;
  const int q0  = rb0 + wave * 16;
  const __bf16* Kq  = kbf  + (size_t)bg * CS * CHD;
  const __bf16* Vv  = vbf  + (size_t)bg * CS * CHD;
  const __bf16* VvT = vbfT + (size_t)bg * CS * CHD;

  Frag16 qf[4];
  {
    const __bf16* qp = Kq + (size_t)(q0 + l16) * CHD + half * 8;
#pragma unroll
    for (int c = 0; c < 4; c++) {
      qf[c].h[0] = *(const bf16x8*)(qp + c * 32);
      qf[c].h[1] = *(const bf16x8*)(qp + c * 32 + 16);
    }
  }

  float m_i[8], l_i[8];
  floatx8 o[8];
  const floatx8 zf = {0.f,0.f,0.f,0.f,0.f,0.f,0.f,0.f};
#pragma unroll
  for (int r = 0; r < 8; r++) { m_i[r] = -1e30f; l_i[r] = 0.f; o[r] = zf; }

  const float scale = 11.313708499f;   // sqrt(128); reference multiplies
  const int nkb = (rb0 + 64) / 32;     // uniform across the block

  for (int kb = 0; kb < nkb; kb++) {
    __syncthreads();
    // async-stage K tile [key][d] and V^T tile [d][key]
#pragma unroll
    for (int t = 0; t < 4; t++) {
      const int c = tid + t * 128;             // 512 chunks
      const int key = c >> 4, ck = c & 15;
      async_copy_b128(&sK[key * 136 + ck * 8],
                      Vv + (size_t)(kb * 32 + key) * CHD + ck * 8);
    }
#pragma unroll
    for (int t = 0; t < 4; t++) {
      const int c = tid + t * 128;             // 512 chunks
      const int d = c >> 2, ck = c & 3;
      async_copy_b128(&sVT[d * 48 + ck * 8],
                      VvT + (size_t)d * CS + kb * 32 + ck * 8);
    }
    wait_async0();
    __syncthreads();

    // scores: two 16-key tiles, K-dim = 128 in 4 chunks
    floatx8 sc[2]; sc[0] = zf; sc[1] = zf;
#pragma unroll
    for (int t = 0; t < 2; t++)
#pragma unroll
      for (int c = 0; c < 4; c++) {
        const __bf16* p = &sK[(t * 16 + l16) * 136 + c * 32 + half * 8];
        Frag16 b; b.h[0] = *(const bf16x8*)p; b.h[1] = *(const bf16x8*)(p + 16);
        sc[t] = __builtin_amdgcn_wmma_f32_16x16x32_bf16(
            false, qf[c].v, false, b.v, (short)0, sc[t], false, false);
      }

    // online softmax update
#pragma unroll
    for (int r = 0; r < 8; r++) {
      const int row = q0 + r + half * 8;
      float rmax = -1e30f;
#pragma unroll
      for (int t = 0; t < 2; t++) {
        const int key = kb * 32 + t * 16 + l16;
        float v = sc[t][r] * scale + ((key <= row) ? 0.f : -1e9f);
        sc[t][r] = v;
        rmax = fmaxf(rmax, v);
      }
      for (int d = 1; d < 16; d <<= 1) rmax = fmaxf(rmax, __shfl_xor(rmax, d, 32));
      const float mnew  = fmaxf(m_i[r], rmax);
      const float alpha = __expf(m_i[r] - mnew);
      m_i[r] = mnew;
      float psum = 0.f;
#pragma unroll
      for (int t = 0; t < 2; t++) {
        const float p = __expf(sc[t][r] - mnew);
        sc[t][r] = p;
        psum += p;
      }
      for (int d = 1; d < 16; d <<= 1) psum += __shfl_xor(psum, d, 32);
      l_i[r] = l_i[r] * alpha + psum;
#pragma unroll
      for (int dt = 0; dt < 8; dt++) o[dt][r] *= alpha;
    }

    // P tile (C-layout) -> LDS -> A-layout fragment
    __bf16* pw = sP + wave * 16 * 48;
#pragma unroll
    for (int t = 0; t < 2; t++)
#pragma unroll
      for (int r = 0; r < 8; r++)
        pw[(r + half * 8) * 48 + t * 16 + l16] = f2bf(sc[t][r]);
    __syncthreads();

    Frag16 pf;
    {
      const __bf16* p = &pw[l16 * 48 + half * 8];
      pf.h[0] = *(const bf16x8*)p;
      pf.h[1] = *(const bf16x8*)(p + 16);
    }

    // attn accumulation: P(16x32) x V(32x128)
#pragma unroll
    for (int dt = 0; dt < 8; dt++) {
      const __bf16* p = &sVT[(dt * 16 + l16) * 48 + half * 8];
      Frag16 b; b.h[0] = *(const bf16x8*)p; b.h[1] = *(const bf16x8*)(p + 16);
      o[dt] = __builtin_amdgcn_wmma_f32_16x16x32_bf16(
          false, pf.v, false, b.v, (short)0, o[dt], false, false);
    }
  }

#pragma unroll
  for (int dt = 0; dt < 8; dt++) {
    const int d = dt * 16 + l16;
#pragma unroll
    for (int r = 0; r < 8; r++) {
      const int row = q0 + r + half * 8;
      attn_out[(size_t)bg * CS * CHD + (size_t)row * CHD + d] = o[dt][r] / l_i[r];
    }
  }
}

// ---------------------------------------------------- gate (sigmoid) + GQA broadcast
__global__ __launch_bounds__(256)
void gate_combine(const float* __restrict__ attn, const float* __restrict__ gate,
                  __bf16* __restrict__ out) {
  const int idx = blockIdx.x * 256 + threadIdx.x;   // (b,s,h,d) = 8388608
  const int d = idx & 127;
  const int h = (idx >> 7) & 15;
  const int s = (idx >> 11) & 2047;
  const int b = idx >> 22;
  const int g = h >> 2;
  const size_t a = ((((size_t)b * CNKV + g) * CS + s) * CHD + d);
  const float gt = gate[idx];
  const float sg = 1.f / (1.f + __expf(-gt));
  out[idx] = f2bf(attn[a] * sg);
}

// ---------------------------------------------------------------- launch
extern "C" void kernel_launch(void* const* d_in, const int* in_sizes, int n_in,
                              void* d_out, int out_size, void* d_ws, size_t ws_size,
                              hipStream_t stream) {
  (void)in_sizes; (void)n_in; (void)out_size; (void)ws_size;
  const float* x    = (const float*)d_in[0];
  const float* cosb = (const float*)d_in[1];
  const float* sinb = (const float*)d_in[2];
  // d_in[3] = attention_mask: causal -1e9; computed analytically in-kernel
  const float* ln1  = (const float*)d_in[4];
  const float* ln2  = (const float*)d_in[5];
  const float* Wq   = (const float*)d_in[6];
  const float* Wk   = (const float*)d_in[7];
  const float* Wv   = (const float*)d_in[8];
  const float* Wo   = (const float*)d_in[9];
  const float* Wg   = (const float*)d_in[10];
  const float* Wu   = (const float*)d_in[11];
  const float* Wd   = (const float*)d_in[12];
  float* out = (float*)d_out;

  float* ws = (float*)d_ws;
  __bf16* h_bf  = (__bf16*)(ws + 0);          //  8388608 bf16
  float*  gate  = ws + 4194304;               //  8388608 f32
  float*  kproj = ws + 12582912;              //  2097152 f32
  float*  vproj = ws + 14680064;              //  2097152 f32
  float*  x1    = ws + 16777216;              //  8388608 f32
  __bf16* h2_bf = (__bf16*)(ws + 25165824);   //  8388608 bf16
  float*  attn  = ws + 29360128;              //  2097152 f32
  __bf16* kbf   = (__bf16*)(ws + 31457280);   //  2097152 bf16
  __bf16* vbf   = (__bf16*)(ws + 32505856);   //  2097152 bf16
  __bf16* vbfT  = (__bf16*)(ws + 33554432);   //  2097152 bf16
  __bf16* gated = (__bf16*)(ws + 34603008);   //  8388608 bf16
  __bf16* u_bf  = (__bf16*)(ws + 38797312);   // 33554432 bf16
  __bf16* mbf   = (__bf16*)(ws + 55574528);   // 33554432 bf16
  __bf16* WqT   = (__bf16*)(ws + 72351744);   // 2048x2048
  __bf16* WkT   = (__bf16*)(ws + 74448896);   //  512x2048
  __bf16* WvT   = (__bf16*)(ws + 74973184);   //  512x2048
  __bf16* WoT   = (__bf16*)(ws + 75497472);   // 2048x2048
  __bf16* WgT   = (__bf16*)(ws + 77594624);   // 8192x2048
  __bf16* WuT   = (__bf16*)(ws + 85983232);   // 8192x2048
  __bf16* WdT   = (__bf16*)(ws + 94371840);   // 2048x8192  (end: 102760448 f32)

  // 0. weights -> bf16 W^T (L2-resident for the GEMMs; K-contiguous rows for async DMA)
  convert_transpose<<<dim3(64,64),  256, 0, stream>>>(Wq, WqT, CHID, CHID, 2*CNH*CHD, CNH*CHD);
  convert_transpose<<<dim3(64,16),  256, 0, stream>>>(Wk, WkT, CHID, CNKV*CHD, CNKV*CHD, 0);
  convert_transpose<<<dim3(64,16),  256, 0, stream>>>(Wv, WvT, CHID, CNKV*CHD, CNKV*CHD, 0);
  convert_transpose<<<dim3(64,64),  256, 0, stream>>>(Wo, WoT, CHID, CHID, CHID, 0);
  convert_transpose<<<dim3(64,256), 256, 0, stream>>>(Wg, WgT, CHID, CFF, CFF, 0);
  convert_transpose<<<dim3(64,256), 256, 0, stream>>>(Wu, WuT, CHID, CFF, CFF, 0);
  convert_transpose<<<dim3(256,64), 256, 0, stream>>>(Wd, WdT, CFF, CHID, CHID, 0);

  // 1. h = rmsnorm(x) * ln1  (bf16)
  rmsnorm_kernel<<<CTOK, 256, 0, stream>>>(x, ln1, h_bf);
  // 2. gate = h @ Wq[:, 2048:]  (first half of Wq is dead in the reference)
  gemm_wmma<0><<<dim3(16,32), 256, 0, stream>>>(h_bf, WqT, gate, nullptr,
      CTOK, CNH*CHD, CHID, CHID, CNH*CHD);
  // 3/4. k,v projections
  gemm_wmma<0><<<dim3(4,32), 256, 0, stream>>>(h_bf, WkT, kproj, nullptr,
      CTOK, CNKV*CHD, CHID, CHID, CNKV*CHD);
  gemm_wmma<0><<<dim3(4,32), 256, 0, stream>>>(h_bf, WvT, vproj, nullptr,
      CTOK, CNKV*CHD, CHID, CHID, CNKV*CHD);
  // 5. rope(v) + pack (b,g,s,d) and transposed (b,g,d,s)
  rope_pack<<<8192, 256, 0, stream>>>(kproj, vproj, cosb, sinb, kbf, vbf, vbfT);
  // 6. flash attention per (b, kv-head): GQA groups identical -> compute once
  flash_attn<<<dim3(CB*CNKV, CS/64), 128, 0, stream>>>(kbf, vbf, vbfT, attn);
  // 7. gated = attn(broadcast over group) * sigmoid(gate)  (bf16)
  gate_combine<<<32768, 256, 0, stream>>>(attn, gate, gated);
  // 8. x1 = gated @ Wo + x  (fused residual)
  gemm_wmma<1><<<dim3(16,32), 256, 0, stream>>>(gated, WoT, x1, x,
      CTOK, CHID, CNH*CHD, CNH*CHD, CHID);
  // 9. h2 = rmsnorm(x1) * ln2  (bf16)
  rmsnorm_kernel<<<CTOK, 256, 0, stream>>>(x1, ln2, h2_bf);
  // 10. u = h2 @ Wu  (bf16 store)
  gemm_wmma<3><<<dim3(64,32), 256, 0, stream>>>(h2_bf, WuT, (void*)u_bf, nullptr,
      CTOK, CFF, CHID, CHID, CFF);
  // 11. m = silu(h2 @ Wg) * u  (fused epilogue, bf16)
  gemm_wmma<2><<<dim3(64,32), 256, 0, stream>>>(h2_bf, WgT, (void*)mbf, u_bf,
      CTOK, CFF, CHID, CHID, CFF);
  // 12. out = m @ Wd + x1  (fused residual, f32 to d_out)
  gemm_wmma<1><<<dim3(16,32), 256, 0, stream>>>(mbf, WdT, out, x1,
      CTOK, CHID, CFF, CFF, CHID);
}
